// RNNModel_35579509080501
// MI455X (gfx1250) — compile-verified, matrix operand order checked by reading, not run on hardware
//
#include <hip/hip_runtime.h>
#include <cstdint>

typedef __attribute__((ext_vector_type(16))) _Float16 v16h;
typedef __attribute__((ext_vector_type(8)))  _Float16 v8h;
typedef __attribute__((ext_vector_type(4)))  _Float16 v4h;
typedef __attribute__((ext_vector_type(2)))  _Float16 v2h;
typedef __attribute__((ext_vector_type(8)))  float    v8f;
typedef __attribute__((ext_vector_type(4)))  unsigned int v4u;
typedef __attribute__((ext_vector_type(8)))  int      v8i;
typedef __attribute__((ext_vector_type(4)))  int      v4i;

constexpr int BATCH  = 2048;
constexpr int TSTEPS = 512;
constexpr int DIN    = 128;
constexpr int H      = 64;
constexpr int NB     = 2;            // 16-row batch tiles per workgroup
constexpr int ROWS   = 16 * NB;      // 32 batch rows per workgroup
constexpr int XPITCH   = DIN + 4;    // f32 staging pitch; matches TDM pad (4 dwords / 128)
constexpr int XPITCH16 = DIN + 8;    // f16 x pitch (136 halves = 272B, stagger banks)
constexpr int HPITCH   = H + 8;      // f16 h pitch  (72 halves = 144B, stagger banks)

#if defined(__HIP_DEVICE_COMPILE__) && __has_builtin(__builtin_amdgcn_tensor_load_to_lds)
#define HAVE_TDM 1
#else
#define HAVE_TDM 0
#endif

// Convert 16 consecutive f32 to a v16h fragment (generic pointer: works for LDS & global).
__device__ __forceinline__ v16h frag16_from_f32(const float* p) {
  v16h r;
#pragma unroll
  for (int q = 0; q < 4; ++q) {
    float4 f = ((const float4*)p)[q];
    r[4*q+0] = (_Float16)f.x; r[4*q+1] = (_Float16)f.y;
    r[4*q+2] = (_Float16)f.z; r[4*q+3] = (_Float16)f.w;
  }
  return r;
}

// Load 16 consecutive f16 from LDS as a v16h fragment (two b128 loads).
__device__ __forceinline__ v16h frag16_from_f16(const _Float16* p) {
  v8h a = *(const v8h*)p;
  v8h b = *(const v8h*)(p + 8);
  return __builtin_shufflevector(a, b, 0,1,2,3,4,5,6,7,8,9,10,11,12,13,14,15);
}

__device__ __forceinline__ v8f wmma_f16(v16h a, v16h b, v8f c) {
  return __builtin_amdgcn_wmma_f32_16x16x32_f16(false, a, false, b, (short)0, c,
                                                false, false);
}

#if HAVE_TDM
// TDM: 2D tile, 32 rows x 128 dwords, row stride T*DIN dwords, +4-dword LDS pad per row.
__device__ __forceinline__ void stage_x_tdm(const float* x, int b0, int t,
                                            unsigned lds_byte_off) {
  uint64_t ga = (uint64_t)(uintptr_t)x + ((uint64_t)b0 * TSTEPS + t) * DIN * 4ull;
  v4u g0 = { 1u,                                   // count=1
             lds_byte_off,                         // lds_addr
             (unsigned)ga,                         // global_addr[31:0]
             ((unsigned)(ga >> 32) & 0x01FFFFFFu) | (2u << 30) }; // addr[56:32], type=2
  v8i g1 = { (int)((2u << 16)       // data_size = 4B
                 | (1u << 20)       // pad_enable
                 | (6u << 22)       // pad_interval: 2^(6+1)=128 dwords
                 | (3u << 25)),     // pad_amount: 4 dwords
             (int)(128u << 16),                 // tensor_dim0 = 128 (lo16 << 16)
             0,                                 // dim0 hi16=0 | dim1 lo16=0 (dim1=1<<20)
             (int)(16u | (128u << 16)),         // dim1 hi16=16 | tile_dim0=128
             (int)32u,                          // tile_dim1=32, tile_dim2=0
             (int)(TSTEPS * DIN),               // tensor_dim0_stride = 65536 dwords
             0, 0 };
  v4i gz = { 0, 0, 0, 0 };
#if __clang_major__ >= 23
  v8i gz8 = { 0, 0, 0, 0, 0, 0, 0, 0 };
  __builtin_amdgcn_tensor_load_to_lds(g0, g1, gz, gz, gz8, 0);
#else
  __builtin_amdgcn_tensor_load_to_lds(g0, g1, gz, gz, 0);
#endif
}
#else
// Fallback: cooperative coalesced global->LDS stage (128 threads x 8 float4).
__device__ __forceinline__ void stage_x_coop(const float* __restrict__ x, int b0, int t,
                                             float* dst, int tid) {
#pragma unroll
  for (int k = 0; k < 8; ++k) {
    int c   = tid + k * 128;      // float4 chunk 0..1023
    int row = c >> 5;
    int c4  = c & 31;
    const float4 v = *((const float4*)(x + ((size_t)(b0 + row) * TSTEPS + t) * DIN) + c4);
    *((float4*)(dst + row * XPITCH) + c4) = v;
  }
}
#endif

__global__ __launch_bounds__(128)
void rnn_fused_kernel(const float* __restrict__ x,
                      const float* __restrict__ Wih, const float* __restrict__ Whh,
                      const float* __restrict__ bih, const float* __restrict__ bhh,
                      const float* __restrict__ fcW, const float* __restrict__ fcb,
                      float* __restrict__ out) {
  __shared__ float    xbuf[2][ROWS][XPITCH];     // TDM landing (f32), double buffered
  __shared__ _Float16 x16 [ROWS][XPITCH16];      // converted x tile (f16)
  __shared__ _Float16 hbuf[2][ROWS][HPITCH];     // hidden state exchange (f16)

  const int tid  = threadIdx.x;
  const int w    = tid >> 5;       // wave id: owns M-tile w (rows 16w..16w+15 of H)
  const int lane = tid & 31;
  const int lq   = lane & 15;      // column (batch) within tile / row within A
  const int g    = lane >> 4;      // lane half: selects K sub-range
  const int b0   = blockIdx.x * ROWS;

  // h0 = 0
  for (int i = tid; i < ROWS * HPITCH; i += 128)
    (&hbuf[0][0][0])[i] = (_Float16)0.f;

  // Resident A fragments: A[m,k] with m = 16w + lq, slot s -> k = 32f + 16g + s.
  v16h aW[4];
  {
    const float* p = Wih + (size_t)(16 * w + lq) * DIN;
#pragma unroll
    for (int f = 0; f < 4; ++f) aW[f] = frag16_from_f32(p + 32 * f + 16 * g);
  }
  v16h aH[2];
  {
    const float* p = Whh + (size_t)(16 * w + lq) * H;
#pragma unroll
    for (int f = 0; f < 2; ++f) aH[f] = frag16_from_f32(p + 32 * f + 16 * g);
  }
  // Bias per D accumulator slot j: m = 16w + 8g + j.
  float biasj[8];
#pragma unroll
  for (int j = 0; j < 8; ++j) {
    int m = 16 * w + 8 * g + j;
    biasj[j] = bih[m] + bhh[m];
  }

  // Prefetch x tile for t = 0.
#if HAVE_TDM
  if (w == 0) stage_x_tdm(x, b0, 0, (unsigned)(uintptr_t)&xbuf[0][0][0]);
#else
  stage_x_coop(x, b0, 0, &xbuf[0][0][0], tid);
#endif

  for (int t = 0; t < TSTEPS; ++t) {
    const int rb = t & 1;
#if HAVE_TDM
    if (w == 0) __builtin_amdgcn_s_wait_tensorcnt(0);
#endif
    __syncthreads();   // x tile t landed; h(t) stores visible; x16/next-buf free

    // Prefetch x tile t+1 into the other buffer.
    if (t + 1 < TSTEPS) {
#if HAVE_TDM
      if (w == 0) stage_x_tdm(x, b0, t + 1, (unsigned)(uintptr_t)&xbuf[(t + 1) & 1][0][0]);
#else
      stage_x_coop(x, b0, t + 1, &xbuf[(t + 1) & 1][0][0], tid);
#endif
    }

    // Cooperative f32 -> f16 convert of tile t (each element converted once).
#pragma unroll
    for (int k = 0; k < 8; ++k) {
      int c   = tid + k * 128;
      int row = c >> 5;
      int c4  = c & 31;
      float4 f = *((const float4*)&xbuf[rb][row][0] + c4);
      v4h h4; h4.x = (_Float16)f.x; h4.y = (_Float16)f.y;
      h4.z = (_Float16)f.z; h4.w = (_Float16)f.w;
      *(v4h*)(&x16[row][0] + c4 * 4) = h4;
    }
    __syncthreads();   // x16 ready

    // D[m,b] = W_ih[m,:]x[b,:] + W_hh[m,:]h[b,:] + bias, relu -> h(t+1)
#pragma unroll
    for (int n = 0; n < NB; ++n) {
      const int r = 16 * n + lq;         // local batch row this lane's column maps to
      v8f acc = {};
#pragma unroll
      for (int f = 0; f < 4; ++f) {
        v16h xb = frag16_from_f16(&x16[r][32 * f + 16 * g]);
        acc = wmma_f16(aW[f], xb, acc);
      }
#pragma unroll
      for (int f = 0; f < 2; ++f) {
        v16h hb = frag16_from_f16(&hbuf[rb][r][32 * f + 16 * g]);
        acc = wmma_f16(aH[f], hb, acc);
      }
      _Float16* hout = &hbuf[rb ^ 1][r][16 * w + 8 * g];
#pragma unroll
      for (int i = 0; i < 4; ++i) {
        float a0 = fmaxf(acc[2 * i]     + biasj[2 * i],     0.f);
        float a1 = fmaxf(acc[2 * i + 1] + biasj[2 * i + 1], 0.f);
        v2h pk; pk.x = (_Float16)a0; pk.y = (_Float16)a1;
        *(v2h*)(hout + 2 * i) = pk;
      }
    }
  }

  __syncthreads();
  // Epilogue: out[b] = sigmoid(h_T[b,:] . fcW + fcb + 1e-6). T even -> final h in hbuf[0].
  if (tid < ROWS) {
    float z = 0.f;
#pragma unroll
    for (int k = 0; k < H; ++k) z += (float)hbuf[0][tid][k] * fcW[k];
    z += fcb[0] + 1e-6f;
    out[b0 + tid] = 1.f / (1.f + __expf(-z));
  }
}

extern "C" void kernel_launch(void* const* d_in, const int* in_sizes, int n_in,
                              void* d_out, int out_size, void* d_ws, size_t ws_size,
                              hipStream_t stream) {
  (void)in_sizes; (void)n_in; (void)out_size; (void)d_ws; (void)ws_size;
  const float* x   = (const float*)d_in[0];
  const float* Wih = (const float*)d_in[1];
  const float* Whh = (const float*)d_in[2];
  const float* bih = (const float*)d_in[3];
  const float* bhh = (const float*)d_in[4];
  const float* fcW = (const float*)d_in[5];
  const float* fcb = (const float*)d_in[6];
  float* out = (float*)d_out;

  dim3 grid(BATCH / ROWS);   // 64 workgroups x 32 batch rows
  dim3 block(128);           // 4 waves: one per M-tile of H
  rnn_fused_kernel<<<grid, block, 0, stream>>>(x, Wih, Whh, bih, bhh, fcW, fcb, out);
}
